// GCNConvolution_1357209666173
// MI455X (gfx1250) — matrix operand order repaired and checked by reading, not compiled, not run on hardware
//
#include <hip/hip_runtime.h>

typedef __attribute__((ext_vector_type(16))) __bf16 v16bf;
typedef __attribute__((ext_vector_type(8)))  float  v8f;

namespace {

constexpr int H = 128;

__global__ __launch_bounds__(256) void zero_kernel(float* __restrict__ p, int n) {
  int i = blockIdx.x * 256 + threadIdx.x;
  if (i < (n >> 2)) reinterpret_cast<float4*>(p)[i] = make_float4(0.f, 0.f, 0.f, 0.f);
}

// OUT = relu(IN @ W + bias) (+ RES if RESIDUAL). IN/W/OUT are f32 in memory;
// matmul runs on v_wmma_f32_16x16x32_bf16 with f32 accumulation.
template <bool RESIDUAL>
__global__ __launch_bounds__(256) void gemm128_relu(
    const float* __restrict__ IN, const float* __restrict__ W,
    const float* __restrict__ bias, const float* __restrict__ RES,
    float* __restrict__ OUT, int nrows) {
  // 32 pre-swizzled B fragments (kc*8+nt), each 32 lanes x 16 bf16 = 1 KB.
  __shared__ __attribute__((aligned(32))) __bf16 smem[32 * 512];

  const int lane = threadIdx.x & 31;
  const int wave = threadIdx.x >> 5;   // 8 waves per block
  const int half = lane >> 4;          // 0: lanes 0-15, 1: lanes 16-31
  const int m    = lane & 15;

  // ---- Stage W (f32 -> bf16) into LDS as ready-made WMMA B fragments ----
  // B layout (32x16 bf16): lane n holds column N=n; lanes 0-15 cover K=kc*32..+15,
  // lanes 16-31 cover K=kc*32+16..+31; VGPR j packs K=2j, 2j+1 of that half.
  #pragma unroll
  for (int ff = 0; ff < 4; ++ff) {
    const int f  = wave + ff * 8;      // fragment id 0..31  ( = kc*8 + nt )
    const int kc = f >> 3;
    const int nt = f & 7;
    __bf16* dst  = &smem[f * 512 + lane * 16];
    const int colw = nt * 16 + m;
    #pragma unroll
    for (int j = 0; j < 8; ++j) {
      const int k0 = kc * 32 + half * 16 + 2 * j;
      dst[2 * j]     = (__bf16)W[(size_t)k0 * H + colw];
      dst[2 * j + 1] = (__bf16)W[(size_t)(k0 + 1) * H + colw];
    }
  }
  __syncthreads();

  const int rowBase = (blockIdx.x * 8 + wave) * 16;
  if (rowBase < nrows) {               // wave-uniform guard: EXEC stays all-ones
    // ---- A fragments: 16 rows x K=128, f32 -> bf16 in-register ----
    // A layout (16x32 bf16): lane m = row M; VGPR j holds K pairs
    // K=2j, 2j+1, +16 for lanes 16-31 (matches B fragments).
    v16bf afrag[4];
    const float* inrow = IN + (size_t)(rowBase + m) * H;
    #pragma unroll
    for (int kc = 0; kc < 4; ++kc) {
      #pragma unroll
      for (int j = 0; j < 8; ++j) {
        const int k0 = kc * 32 + half * 16 + 2 * j;
        const float2 v = *reinterpret_cast<const float2*>(inrow + k0);
        afrag[kc][2 * j]     = (__bf16)v.x;
        afrag[kc][2 * j + 1] = (__bf16)v.y;
      }
    }

    #pragma unroll
    for (int nt = 0; nt < 8; ++nt) {
      v8f acc = {};
      #pragma unroll
      for (int kc = 0; kc < 4; ++kc) {
        const v16bf bfrag =
            *reinterpret_cast<const v16bf*>(&smem[(kc * 8 + nt) * 512 + lane * 16]);
        acc = __builtin_amdgcn_wmma_f32_16x16x32_bf16(
            false, afrag[kc], false, bfrag, (short)0, acc, false, false);
      }
      // C/D layout: VGPR i holds row rowBase + half*8 + i, column nt*16 + m.
      const int col  = nt * 16 + m;
      const float bv = bias[col];
      #pragma unroll
      for (int i = 0; i < 8; ++i) {
        const int row = rowBase + half * 8 + i;
        float v = acc[i] + bv;
        v = v > 0.f ? v : 0.f;
        if (RESIDUAL) v += RES[(size_t)row * H + col];
        OUT[(size_t)row * H + col] = v;
      }
    }
  }
}

// pooled[tgt[e], :] += gcn_norm[e] * y[src[e], :]   (one wave per edge,
// float4 per lane, relaxed agent-scope f32 atomics -> L2-side atomic units)
__global__ __launch_bounds__(256) void edge_scatter(
    const float* __restrict__ y, const int* __restrict__ src,
    const int* __restrict__ tgt, const float* __restrict__ norm,
    float* __restrict__ pooled, int nedges) {
  const int t = blockIdx.x * 256 + threadIdx.x;
  const int e = t >> 5;
  if (e >= nedges) return;
  const int lane = t & 31;
  const int s = src[e];
  const int d = tgt[e];
  const float g = norm[e];
  const float4 v = *reinterpret_cast<const float4*>(y + (size_t)s * H + lane * 4);
  float* p = pooled + (size_t)d * H + lane * 4;
  __hip_atomic_fetch_add(p + 0, g * v.x, __ATOMIC_RELAXED, __HIP_MEMORY_SCOPE_AGENT);
  __hip_atomic_fetch_add(p + 1, g * v.y, __ATOMIC_RELAXED, __HIP_MEMORY_SCOPE_AGENT);
  __hip_atomic_fetch_add(p + 2, g * v.z, __ATOMIC_RELAXED, __HIP_MEMORY_SCOPE_AGENT);
  __hip_atomic_fetch_add(p + 3, g * v.w, __ATOMIC_RELAXED, __HIP_MEMORY_SCOPE_AGENT);
}

}  // namespace

extern "C" void kernel_launch(void* const* d_in, const int* in_sizes, int n_in,
                              void* d_out, int out_size, void* d_ws, size_t ws_size,
                              hipStream_t stream) {
  const float* x      = (const float*)d_in[0];
  const int*   src    = (const int*)d_in[1];
  const int*   tgt    = (const int*)d_in[2];
  const float* gnorm  = (const float*)d_in[3];
  const float* W_edge = (const float*)d_in[4];
  const float* b_edge = (const float*)d_in[5];
  const float* W_node = (const float*)d_in[6];
  const float* b_node = (const float*)d_in[7];
  float* out = (float*)d_out;

  const int nodes = in_sizes[0] / H;  // 50000
  const int edges = in_sizes[1];      // 625000

  float* y      = (float*)d_ws;               // [nodes, H] f32 (25.6 MB)
  float* pooled = y + (size_t)nodes * H;      // [nodes, H] f32 (25.6 MB)

  const int rowTiles   = (nodes + 15) / 16;
  const int gemmBlocks = (rowTiles + 7) / 8;

  // 1) Hoisted edge transform: y = relu(x @ W_edge + b_edge)   (node-level!)
  gemm128_relu<false><<<gemmBlocks, 256, 0, stream>>>(x, W_edge, b_edge, nullptr, y, nodes);

  // 2) pooled = 0 (must re-zero every replay)
  const int n = nodes * H;
  zero_kernel<<<((n >> 2) + 255) / 256, 256, 0, stream>>>(pooled, n);

  // 3) pooled[tgt] += gcn_norm * y[src]   (L2-resident gather + atomics)
  edge_scatter<<<(edges * 32 + 255) / 256, 256, 0, stream>>>(y, src, tgt, gnorm, pooled, edges);

  // 4) out = relu(pooled @ W_node + b_node) + x
  gemm128_relu<true><<<gemmBlocks, 256, 0, stream>>>(pooled, W_node, b_node, x, out, nodes);
}
